// NeuralCDE_stacked_10617159155882
// MI455X (gfx1250) — matrix operand order repaired and checked by reading, not compiled, not run on hardware
//
#include <hip/hip_runtime.h>
#include <hip/hip_bf16.h>
#include <math.h>

// Problem dims
#define B_  512
#define S_  128
#define C_  12
#define H_  64
#define O_  20

typedef __attribute__((ext_vector_type(16))) __bf16 v16bf;
typedef __attribute__((ext_vector_type(8)))  float  v8f;

union FragU { uint4 q[2]; v16bf v; };

// A-operand fragment (16x32 bf16 slice of a row-major [16][64] matrix).
// Per ISA 7.12.2: lane m=L&15; half=L>>4; two contiguous 8-elem K runs at
// k0+half*8 and k0+16+half*8.
__device__ __forceinline__ v16bf load_fragA(const __bf16* rowbase, int k0, int half) {
  FragU f;
  const uint4* p = (const uint4*)rowbase;
  f.q[0] = p[(k0 + half * 8) >> 3];
  f.q[1] = p[(k0 + 16 + half * 8) >> 3];
  return f.v;
}
// B-operand fragment (32x16 bf16) from transposed storage [n][64]:
// lane n=L&15; two contiguous 8-elem K runs at k0+half*8 and k0+16+half*8.
__device__ __forceinline__ v16bf load_fragB(const __bf16* rowbase, int k0, int half) {
  FragU f;
  const uint4* p = (const uint4*)rowbase;
  f.q[0] = p[(k0 + half * 8) >> 3];
  f.q[1] = p[(k0 + 16 + half * 8) >> 3];
  return f.v;
}

#define WMMA_BF16(A, Bm, Cm) \
  __builtin_amdgcn_wmma_f32_16x16x32_bf16(false, (A), false, (Bm), (short)0, (Cm), false, false)

// Async copy of one 16x64 bf16 weight tile (2 KB) global -> LDS.
// 4 x GLOBAL_LOAD_ASYNC_TO_LDS_B128: 64 B per lane, tracked by ASYNCcnt.
__device__ __forceinline__ void async_copy_tile(const __bf16* gbase, __bf16* lbuf, int lane) {
  unsigned           laddr = (unsigned)(uintptr_t)lbuf + (unsigned)lane * 64u;
  unsigned long long gaddr = (unsigned long long)(uintptr_t)gbase + (unsigned long long)lane * 64ull;
  asm volatile("global_load_async_to_lds_b128 %0, %1, off\n\t"
               "global_load_async_to_lds_b128 %0, %1, off offset:16\n\t"
               "global_load_async_to_lds_b128 %0, %1, off offset:32\n\t"
               "global_load_async_to_lds_b128 %0, %1, off offset:48"
               :: "v"(laddr), "v"(gaddr) : "memory");
}
__device__ __forceinline__ void wait_async4() {
  asm volatile("s_wait_asynccnt 0x4" ::: "memory");
}
__device__ __forceinline__ void wait_async0() {
  asm volatile("s_wait_asynccnt 0x0" ::: "memory");
}

// Prep: transpose+convert f2_W2 (64 x 4096 f32, row-major [k][n]) -> ws bf16 [n][64]
__global__ void prep_w2(const float* __restrict__ src, __bf16* __restrict__ dst) {
  int i = blockIdx.x * blockDim.x + threadIdx.x;     // 4096*64 = 262144 threads
  if (i < 4096 * 64) {
    int n = i >> 6, k = i & 63;
    dst[i] = (__bf16)src[k * 4096 + n];
  }
}

__global__ void __launch_bounds__(256, 1)
cde_kernel(const float* __restrict__ coeffs,
           const float* __restrict__ W_init, const float* __restrict__ b_init,
           const float* __restrict__ f1_W1,  const float* __restrict__ f1_b1,
           const float* __restrict__ f1_W2,  const float* __restrict__ f1_b2,
           const float* __restrict__ f2_W1,  const float* __restrict__ f2_b1,
           const float* __restrict__ f2_b2,
           const float* __restrict__ W_ro,   const float* __restrict__ b_ro,
           const __bf16* __restrict__ w2_2T,            // [4096][64] bf16 in d_ws
           float* __restrict__ out)
{
  extern __shared__ char smem[];
  // ---- LDS carve (all uint4-aligned) : total 196864 bytes ----
  __bf16* W2_1T = (__bf16*)smem;                     // [768][64]  96 KB
  __bf16* W1_1T = W2_1T + 768 * 64;                  // [64][64]    8 KB
  __bf16* W1_2T = W1_1T + 64 * 64;                   // [64][64]    8 KB
  __bf16* z1b   = W1_2T + 64 * 64;                   // [16][64]    2 KB
  __bf16* z2b   = z1b + 16 * 64;
  __bf16* h1b   = z2b + 16 * 64;
  __bf16* h2b   = h1b + 16 * 64;
  float* b1_1  = (float*)(h2b + 16 * 64);            // 64
  float* b2_1  = b1_1 + 64;                          // 768
  float* b1_2  = b2_1 + 768;                         // 64
  float* b2_2  = b1_2 + 64;                          // 4096
  float* z1f   = b2_2 + 4096;                        // 1024
  float* z2f   = z1f + 1024;                         // 1024
  float* z1acc = z2f + 1024;                         // 1024
  float* z2new = z1acc + 1024;                       // 1024
  float* dzs   = z2new + 1024;                       // 1024
  float* dxs   = dzs + 1024;                         // 192
  __bf16* ebuf = (__bf16*)(dxs + 192);               // 8 waves x 2 bufs x 1024 = 32 KB

  const int tid = threadIdx.x;
  const int m0  = blockIdx.x * 16;                   // batch tile
  const int lane = tid & 31;
  const int wv   = tid >> 5;                         // 8 waves
  const int half = lane >> 4;
  const int ln   = lane & 15;
  __bf16* mybuf = ebuf + wv * 2048;                  // two 1024-elem buffers

  // ---- one-time: weights -> LDS (bf16, transposed [n][k]) ----
  for (int i = tid; i < 768 * 64; i += 256) {
    int n = i >> 6, k = i & 63;
    W2_1T[i] = (__bf16)f1_W2[k * 768 + n];
  }
  for (int i = tid; i < 64 * 64; i += 256) {
    int n = i >> 6, k = i & 63;
    W1_1T[i] = (__bf16)f1_W1[k * 64 + n];
    W1_2T[i] = (__bf16)f2_W1[k * 64 + n];
  }
  for (int i = tid; i < 64; i += 256)   { b1_1[i] = f1_b1[i]; b1_2[i] = f2_b1[i]; }
  for (int i = tid; i < 768; i += 256)  b2_1[i] = f1_b2[i];
  for (int i = tid; i < 4096; i += 256) b2_2[i] = f2_b2[i];

  // ---- z0 = coeffs[:,0] @ W_init + b_init ; z1 = z2 = z0 ----
  for (int i = tid; i < 1024; i += 256) {
    int m = i >> 6, h = i & 63;
    float s = b_init[h];
    const float* cr = coeffs + (size_t)(m0 + m) * S_ * C_;
#pragma unroll
    for (int c = 0; c < C_; ++c) s += cr[c] * W_init[c * H_ + h];
    z1f[i] = s; z2f[i] = s;
    __bf16 bb = (__bf16)s;
    z1b[i] = bb; z2b[i] = bb;
  }
  __syncthreads();

  // ================= fused time loop (127 steps) =================
  for (int t = 0; t < S_ - 1; ++t) {
    // --- stage A: dx = x_{t+1}-x_t ; z1acc <- z1 ---
    if (tid < 16 * C_) {
      int m = tid / C_, c = tid % C_;
      const float* cp = coeffs + ((size_t)(m0 + m) * S_ + t) * C_ + c;
      dxs[tid] = cp[C_] - cp[0];
    }
    for (int i = tid; i < 1024; i += 256) z1acc[i] = z1f[i];
    __syncthreads();

    // --- stage B: h1 = relu(z1@W1_1+b) [waves 0-3], h2 = relu(z2@W1_2+b) [waves 4-7] ---
    {
      const __bf16* src  = (wv < 4) ? z1b : z2b;
      const __bf16* wt   = (wv < 4) ? W1_1T : W1_2T;
      const float*  bias = (wv < 4) ? b1_1 : b1_2;
      __bf16*       dst  = (wv < 4) ? h1b : h2b;
      const int n0 = (wv & 3) * 16;
      v16bf a0 = load_fragA(src + ln * 64, 0, half);
      v16bf a1 = load_fragA(src + ln * 64, 32, half);
      v16bf bm0 = load_fragB(wt + (n0 + ln) * 64, 0, half);
      v16bf bm1 = load_fragB(wt + (n0 + ln) * 64, 32, half);
      v8f c = {};
      c = WMMA_BF16(a0, bm0, c);
      c = WMMA_BF16(a1, bm1, c);
      const float bv = bias[n0 + ln];
#pragma unroll
      for (int r = 0; r < 8; ++r) {
        float v = c[r] + bv;
        v = v > 0.f ? v : 0.f;
        dst[(r + 8 * half) * 64 + (n0 + ln)] = (__bf16)v;
      }
    }
    __syncthreads();

    // --- stage C: A1 = tanh(h1@W2_1+b), contract with dx -> z1acc (48 tiles, 6/wave) ---
    {
      v16bf a0 = load_fragA(h1b + ln * 64, 0, half);
      v16bf a1 = load_fragA(h1b + ln * 64, 32, half);
      for (int tt = wv * 6; tt < wv * 6 + 6; ++tt) {
        const int ng = tt * 16 + ln;                 // global N index this lane owns
        v16bf bm0 = load_fragB(W2_1T + ng * 64, 0, half);
        v16bf bm1 = load_fragB(W2_1T + ng * 64, 32, half);
        v8f c = {};
        c = WMMA_BF16(a0, bm0, c);
        c = WMMA_BF16(a1, bm1, c);
        const int h = ng / C_, cc = ng - h * C_;
        const float bv = b2_1[ng];
#pragma unroll
        for (int r = 0; r < 8; ++r) {
          const int m = r + 8 * half;
          float v = tanhf(c[r] + bv);
          atomicAdd(&z1acc[m * 64 + h], v * dxs[m * C_ + cc]);
        }
      }
    }
    __syncthreads();

    // --- stage D: dz = z1_new - z1 ; commit z1 ---
    for (int i = tid; i < 1024; i += 256) {
      float nz = z1acc[i];
      dzs[i] = nz - z1f[i];
      z1f[i] = nz;
      z1b[i] = (__bf16)nz;
    }
    __syncthreads();

    // --- stage E: A2 = tanh(h2@W2_2+b), contract with dz -> z2new ---
    // 64 h-groups (64 N each), 8 per wave. W2_2 tiles (16x64 bf16 = 2 KB)
    // streamed global->LDS with double-buffered GLOBAL_LOAD_ASYNC_TO_LDS_B128.
    {
      v16bf a0 = load_fragA(h2b + ln * 64, 0, half);
      v16bf a1 = load_fragA(h2b + ln * 64, 32, half);
      // prologue: kick tile 0
      async_copy_tile(w2_2T + (size_t)(wv * 8) * 64 * 64, mybuf, lane);
      int nb = 0;
      for (int hgl = 0; hgl < 8; ++hgl) {
        const int hg = wv * 8 + hgl;
        float acc[8];
#pragma unroll
        for (int r = 0; r < 8; ++r) acc[r] = 0.f;
#pragma unroll
        for (int sub = 0; sub < 4; ++sub) {
          const int nidx = hgl * 4 + sub + 1;        // next tile index for this wave
          if (nidx < 32) {
            const int nhg = wv * 8 + (nidx >> 2), nsub = nidx & 3;
            async_copy_tile(w2_2T + (size_t)(nhg * 64 + nsub * 16) * 64,
                            mybuf + (nb ^ 1) * 1024, lane);
            wait_async4();                            // tile (hg,sub) has landed
          } else {
            wait_async0();                            // last tile: drain
          }
          const __bf16* tile = mybuf + nb * 1024;     // 16 rows x 64 K, row = local n
          v16bf bm0 = load_fragB(tile + ln * 64, 0, half);
          v16bf bm1 = load_fragB(tile + ln * 64, 32, half);
          v8f c = {};
          c = WMMA_BF16(a0, bm0, c);
          c = WMMA_BF16(a1, bm1, c);
          const int ng = hg * 64 + sub * 16 + ln;
          const float bv = b2_2[ng];
          const int j = sub * 16 + ln;
#pragma unroll
          for (int r = 0; r < 8; ++r) {
            const int m = r + 8 * half;
            float v = tanhf(c[r] + bv);
            acc[r] += v * dzs[m * 64 + j];
          }
          nb ^= 1;
        }
#pragma unroll
        for (int r = 0; r < 8; ++r) {                // reduce over 16 N-lanes
          float s = acc[r];
          s += __shfl_xor(s, 1, 32);
          s += __shfl_xor(s, 2, 32);
          s += __shfl_xor(s, 4, 32);
          s += __shfl_xor(s, 8, 32);
          if (ln == 0) {
            const int m = r + 8 * half;
            z2new[m * 64 + hg] = z2f[m * 64 + hg] + s;
          }
        }
      }
    }
    __syncthreads();

    // --- stage F: commit z2 ---
    for (int i = tid; i < 1024; i += 256) {
      float nz = z2new[i];
      z2f[i] = nz;
      z2b[i] = (__bf16)nz;
    }
    __syncthreads();
  }

  // ---- readout: out = z2 @ W_ro + b_ro ----
  for (int i = tid; i < 16 * O_; i += 256) {
    int m = i / O_, o = i % O_;
    float s = b_ro[o];
#pragma unroll
    for (int h = 0; h < H_; ++h) s += z2f[m * 64 + h] * W_ro[h * O_ + o];
    out[(size_t)(m0 + m) * O_ + o] = s;
  }
}

extern "C" void kernel_launch(void* const* d_in, const int* in_sizes, int n_in,
                              void* d_out, int out_size, void* d_ws, size_t ws_size,
                              hipStream_t stream) {
  const float* coeffs = (const float*)d_in[0];
  // d_in[1] = times (unused by the math)
  const float* W_init = (const float*)d_in[2];
  const float* b_init = (const float*)d_in[3];
  const float* f1_W1  = (const float*)d_in[4];
  const float* f1_b1  = (const float*)d_in[5];
  const float* f1_W2  = (const float*)d_in[6];
  const float* f1_b2  = (const float*)d_in[7];
  const float* f2_W1  = (const float*)d_in[8];
  const float* f2_b1  = (const float*)d_in[9];
  const float* f2_W2  = (const float*)d_in[10];
  const float* f2_b2  = (const float*)d_in[11];
  const float* W_ro   = (const float*)d_in[12];
  const float* b_ro   = (const float*)d_in[13];
  float* out = (float*)d_out;
  __bf16* w2t = (__bf16*)d_ws;                       // 4096*64 bf16 = 512 KB

  prep_w2<<<1024, 256, 0, stream>>>(f2_W2, w2t);

  const int smem = 196864;
  (void)hipFuncSetAttribute((const void*)cde_kernel,
                            hipFuncAttributeMaxDynamicSharedMemorySize, smem);
  cde_kernel<<<B_ / 16, 256, smem, stream>>>(coeffs, W_init, b_init,
                                             f1_W1, f1_b1, f1_W2, f1_b2,
                                             f2_W1, f2_b1, f2_b2,
                                             W_ro, b_ro, w2t, out);
}
